// Mamba2DecoderLayer_26525718020395
// MI455X (gfx1250) — compile-verified, matrix operand order checked by reading, not run on hardware
//
#include <hip/hip_runtime.h>
#include <hip/hip_bf16.h>
#include <math.h>

#if __has_builtin(__builtin_amdgcn_global_load_async_to_lds_b128)
#define USE_ASYNC_LDS 1
#else
#define USE_ASYNC_LDS 0
#endif

namespace {

constexpr int BDIM = 2;
constexpr int LDIM = 2048;
constexpr int DDIM = 2048;
constexpr int DIN_ = 4096;
constexpr int NST  = 128;
constexpr int HH   = 64;
constexpr int PH   = 64;
constexpr int KC   = 4;
constexpr int CONVD = DIN_ + 2*NST;        // 4352
constexpr int NPROJ = 2*DIN_ + 2*NST + HH; // 8512
constexpr int QC   = 64;
constexpr int NCH  = LDIM/QC;              // 32
constexpr int ROWS = BDIM*LDIM;            // 4096
constexpr float EPSV = 1e-5f;
constexpr int SR = 136;  // LDS row stride for 128-wide K rows (pad 8, 16B aligned)
constexpr int SQ = 72;   // LDS row stride for  64-wide K rows (pad 8, 16B aligned)

typedef __bf16 bf16t;
typedef __attribute__((ext_vector_type(16))) __bf16 v16bf;
typedef __attribute__((ext_vector_type(8)))  __bf16 v8bf;
typedef __attribute__((ext_vector_type(8)))  float  v8f;
typedef int v4i __attribute__((ext_vector_type(4)));

#if USE_ASYNC_LDS
typedef __attribute__((address_space(1))) v4i* as1_v4i;
typedef __attribute__((address_space(3))) v4i* as3_v4i;
#endif

__device__ __forceinline__ v8f wmma_bf(const v16bf& a, const v16bf& b, const v8f& c){
  return __builtin_amdgcn_wmma_f32_16x16x32_bf16(false, a, false, b, (short)0, c, false, false);
}

// Load one 16-element bf16 WMMA fragment from an LDS row that stores K
// contiguously: two 16B chunks at {klo, klo+16}, klo = (lane>=16 ? 8 : 0),
// matching the ISA 7.12.2 K-grouping {0..7,16..23} / {8..15,24..31}.
__device__ __forceinline__ v16bf frag_ld(const bf16t* p){
  v8bf x0 = *reinterpret_cast<const v8bf*>(p);
  v8bf x1 = *reinterpret_cast<const v8bf*>(p + 16);
  v16bf r;
  #pragma unroll
  for (int i=0;i<8;i++){ r[i]=x0[i]; r[8+i]=x1[i]; }
  return r;
}

__device__ __forceinline__ float siluf(float x){ return x / (1.f + __expf(-x)); }

__device__ __forceinline__ void wait_async0(){
#if __has_builtin(__builtin_amdgcn_s_wait_asynccnt)
  __builtin_amdgcn_s_wait_asynccnt(0);
#else
  asm volatile("s_wait_asynccnt 0x0" ::: "memory");
#endif
}

// ---------------------------------------------------------------- fp32 -> bf16
__global__ void k_f2bf(const float* __restrict__ src, bf16t* __restrict__ dst, int n){
  for (int i = blockIdx.x*blockDim.x + threadIdx.x; i < n; i += gridDim.x*blockDim.x)
    dst[i] = (bf16t)src[i];
}

// -------------------------------------- resid = hs + residual ; h = rmsnorm(resid)
__global__ __launch_bounds__(256)
void k_fuse_norm(const float* __restrict__ hs, const float* __restrict__ res,
                 const float* __restrict__ w, float* __restrict__ resid_out,
                 bf16t* __restrict__ h_bf){
  int row = blockIdx.x;
  const float* a = hs  + (size_t)row*DDIM;
  const float* b = res + (size_t)row*DDIM;
  float* ro = resid_out + (size_t)row*DDIM;
  __shared__ float red[256];
  float loc[8]; float ss = 0.f;
  #pragma unroll
  for (int i=0;i<8;i++){
    int c = threadIdx.x + i*256;
    float v = a[c] + b[c];
    loc[i] = v; ro[c] = v; ss += v*v;
  }
  red[threadIdx.x] = ss; __syncthreads();
  for (int s=128;s>0;s>>=1){ if (threadIdx.x < s) red[threadIdx.x] += red[threadIdx.x+s]; __syncthreads(); }
  float rstd = rsqrtf(red[0]/(float)DDIM + EPSV);
  #pragma unroll
  for (int i=0;i<8;i++){
    int c = threadIdx.x + i*256;
    h_bf[(size_t)row*DDIM + c] = (bf16t)(loc[i]*rstd*w[c]);
  }
}

// ----------------------------------------------------- bf16 WMMA GEMM C=A@B (f32)
// A: MxK row-major bf16 (M multiple of 128, K multiple of 32),
// Bw: KxN row-major bf16 (N multiple of 8), C: MxN f32.
// Block tile 128x128, double-buffered LDS, async prefetch of tile k+1
// overlapped with WMMA on tile k (one barrier per k-step).
__global__ __launch_bounds__(256)
void k_gemm_bf16(const bf16t* __restrict__ A, const bf16t* __restrict__ Bw,
                 float* __restrict__ C, int M, int N, int K){
  __shared__ __align__(16) bf16t As[2][128*40];  // row-major M x K-tile, stride 40
  __shared__ __align__(16) bf16t Bt[2][128*40];  // K-transposed [n][k], stride 40
  int tid = threadIdx.x, lane = tid & 31, w = tid >> 5;
  int l15 = lane & 15;
  int m0 = blockIdx.y*128, n0 = blockIdx.x*128;
  int wm = (w>>1)*32, wn = (w&1)*64;
  v8f acc[2][4] = {};

  auto fill = [&](int k0, int buf){
    #pragma unroll
    for (int r=0;r<2;r++){ // A tile: 512 x 16B chunks, 2 per thread, async -> LDS
      int ch = tid + 256*r;
      int row = ch>>2, c8 = (ch&3)*8;
      const bf16t* src = A + (size_t)(m0+row)*K + k0 + c8;
#if USE_ASYNC_LDS
      __builtin_amdgcn_global_load_async_to_lds_b128(
          (as1_v4i)(unsigned long long)src,
          (as3_v4i)(unsigned)(unsigned long long)&As[buf][row*40 + c8], 0, 0);
#else
      *reinterpret_cast<v8bf*>(&As[buf][row*40 + c8]) =
          *reinterpret_cast<const v8bf*>(src);
#endif
    }
    #pragma unroll
    for (int r=0;r<2;r++){ // B tile, stored K-transposed; per-chunk guard only
      int kk = (tid>>4) + r*16, n8 = (tid&15)*8;
      v8bf v = {};
      if (n0 + n8 < N)
        v = *reinterpret_cast<const v8bf*>(Bw + (size_t)(k0+kk)*N + n0 + n8);
      #pragma unroll
      for (int i=0;i<8;i++) Bt[buf][(n8+i)*40 + kk] = v[i];
    }
  };

  fill(0, 0);
  int cur = 0;
  for (int k0 = 0; k0 < K; k0 += 32){
#if USE_ASYNC_LDS
    wait_async0();                 // retire async fills for buffer `cur`
#endif
    __syncthreads();               // all fills for `cur` visible block-wide
    if (k0 + 32 < K) fill(k0 + 32, cur ^ 1);   // prefetch next tile during compute
    int klo = (lane>>4)*8;
    v16bf af[2], bfr[4];
    #pragma unroll
    for (int mt=0;mt<2;mt++)
      af[mt] = frag_ld(&As[cur][(wm + mt*16 + l15)*40 + klo]);
    #pragma unroll
    for (int nt=0;nt<4;nt++)
      bfr[nt] = frag_ld(&Bt[cur][(wn + nt*16 + l15)*40 + klo]);
    #pragma unroll
    for (int mt=0;mt<2;mt++)
      #pragma unroll
      for (int nt=0;nt<4;nt++)
        acc[mt][nt] = wmma_bf(af[mt], bfr[nt], acc[mt][nt]);
    cur ^= 1;
  }
  #pragma unroll
  for (int mt=0;mt<2;mt++)
    #pragma unroll
    for (int nt=0;nt<4;nt++)
      #pragma unroll
      for (int v=0;v<8;v++){
        int rr = m0 + wm + mt*16 + v + ((lane>>4)<<3);
        int cc = n0 + wn + nt*16 + l15;
        if (cc < N) C[(size_t)rr*N + cc] = acc[mt][nt][v];
      }
}

// ------------------------- causal depthwise conv(K=4)+bias+SiLU ; dt=softplus(dt+b)
__global__ __launch_bounds__(256)
void k_conv_silu(const float* __restrict__ proj, const float* __restrict__ cw,
                 const float* __restrict__ cb, const float* __restrict__ dtb,
                 float* __restrict__ xBCs, float* __restrict__ dts){
  int row = blockIdx.x;            // b*L + l
  int l = row % LDIM;
  for (int ch = threadIdx.x; ch < CONVD; ch += 256){
    float s = cb[ch];
    #pragma unroll
    for (int k=0;k<KC;k++){
      int lk = l + k - (KC-1);
      if (lk >= 0)
        s += cw[ch*KC + k] * proj[(size_t)(row + k - (KC-1))*NPROJ + DIN_ + ch];
    }
    xBCs[(size_t)row*CONVD + ch] = siluf(s);
  }
  if (threadIdx.x < HH){
    float v = proj[(size_t)row*NPROJ + DIN_ + CONVD + threadIdx.x] + dtb[threadIdx.x];
    dts[(size_t)row*HH + threadIdx.x] = (v > 20.f) ? v : log1pf(expf(v));
  }
}

// --------------------------------------------------- SSD chunked scan per (b,h)
// All WMMA operands staged in LDS with K contiguous per row -> vector fragments.
__global__ __launch_bounds__(256)
void k_ssd(const float* __restrict__ xBCs, const float* __restrict__ dts,
           const float* __restrict__ A_log, const float* __restrict__ D_skip,
           float* __restrict__ y_raw){
  int bh = blockIdx.x;
  int b = bh / HH, h = bh % HH;
  int tid = threadIdx.x, lane = tid & 31, w = tid >> 5;
  int l15 = lane & 15;
  int klo = (lane >> 4) * 8;
  __shared__ __align__(16) bf16t Ccb [QC*SR];   // C (q,n); scaled by e^acs after CB
  __shared__ __align__(16) bf16t Bcb [QC*SR];   // B (q,n)
  __shared__ __align__(16) bf16t BdT [NST*SQ];  // dec[q]*B  stored (n,q)
  __shared__ __align__(16) bf16t xqbT[PH*SQ];   // x*dt      stored (p,q)
  __shared__ __align__(16) bf16t Mb  [QC*SQ];   // masked CB*L (q,s)
  __shared__ __align__(16) bf16t Sb  [PH*SR];   // bf16 snapshot of S (p,n)
  __shared__ float S[PH*NST];                   // running state f32 (p,n)
  __shared__ float dtc[QC], acs[QC], eacs[QC], dec[QC];
  __shared__ float sumA;
  float Ah = -__expf(A_log[h]);
  float Dh = D_skip[h];
  for (int i = tid; i < PH*NST; i += 256) S[i] = 0.f;

  for (int c = 0; c < NCH; c++){
    __syncthreads();
    // -- stage 1: load B,C for this chunk; dt per row
    for (int i = tid; i < QC*NST; i += 256){
      int q = i / NST, n = i % NST;
      size_t rb = (size_t)(b*LDIM + c*QC + q)*CONVD;
      Bcb[q*SR + n] = (bf16t)xBCs[rb + DIN_ + n];
      Ccb[q*SR + n] = (bf16t)xBCs[rb + DIN_ + NST + n];
    }
    if (tid < QC) dtc[tid] = dts[(size_t)(b*LDIM + c*QC + tid)*HH + h];
    __syncthreads();
    // -- stage 2: xq transposed (p,q); serial cumsum of a = dt*A
    for (int i = tid; i < QC*PH; i += 256){
      int q = i / PH, p = i % PH;
      float xs = xBCs[(size_t)(b*LDIM + c*QC + q)*CONVD + h*PH + p];
      xqbT[p*SQ + q] = (bf16t)(xs * dtc[q]);
    }
    if (tid == 0){
      float run = 0.f;
      for (int q=0;q<QC;q++){ run += dtc[q]*Ah; acs[q] = run; eacs[q] = __expf(run); }
      sumA = run;
      for (int q=0;q<QC;q++) dec[q] = __expf(run - acs[q]);
    }
    __syncthreads();

    // -- CB = C @ B^T (64x64, K=n=128): A row q of Ccb, B row s of Bcb
    #pragma unroll
    for (int tt=0;tt<2;tt++){
      int t = w + tt*8, tm = (t>>2)*16, tn = (t&3)*16;
      v8f acc = {};
      for (int kb=0; kb<NST; kb+=32)
        acc = wmma_bf(frag_ld(&Ccb[(tm+l15)*SR + kb + klo]),
                      frag_ld(&Bcb[(tn+l15)*SR + kb + klo]), acc);
      #pragma unroll
      for (int v=0;v<8;v++){
        int q = tm + v + ((lane>>4)<<3);
        int s = tn + l15;
        float val = (s <= q) ? acc[v]*__expf(acs[q]-acs[s]) : 0.f;
        Mb[q*SQ + s] = (bf16t)val;
      }
    }
    __syncthreads();

    // -- stage 5: prescale C by e^acs (C dead after CB); build dec*B transposed;
    //            snapshot S as bf16
    for (int i = tid; i < QC*NST; i += 256){
      int q = i / NST, n = i % NST;
      Ccb[q*SR + n] = (bf16t)(eacs[q]*(float)Ccb[q*SR + n]);
      BdT[n*SQ + q] = (bf16t)(dec[q] *(float)Bcb[q*SR + n]);
    }
    for (int i = tid; i < PH*NST; i += 256){
      int p = i / NST, n = i % NST;
      Sb[p*SR + n] = (bf16t)S[p*NST + n];
    }
    __syncthreads();

    // -- Y = Mb @ xq + Ccb(scaled) @ Sb^T ; write out with D-skip
    #pragma unroll
    for (int tt=0;tt<2;tt++){
      int t = w + tt*8, tm = (t>>2)*16, tn = (t&3)*16;
      v8f acc = {};
      for (int kb=0; kb<QC; kb+=32)            // Y_diag: K = s
        acc = wmma_bf(frag_ld(&Mb  [(tm+l15)*SQ + kb + klo]),
                      frag_ld(&xqbT[(tn+l15)*SQ + kb + klo]), acc);
      for (int kb=0; kb<NST; kb+=32)           // Y_off : K = n
        acc = wmma_bf(frag_ld(&Ccb[(tm+l15)*SR + kb + klo]),
                      frag_ld(&Sb [(tn+l15)*SR + kb + klo]), acc);
      #pragma unroll
      for (int v=0;v<8;v++){
        int q = tm + v + ((lane>>4)<<3);
        int p = tn + l15;
        size_t row = (size_t)(b*LDIM + c*QC + q);
        float xs = xBCs[row*CONVD + h*PH + p];
        y_raw[row*DIN_ + h*PH + p] = acc[v] + Dh*xs;
      }
    }

    // -- states = xq^T @ (dec*B) ; S = e^{sumA}*S + states
    // (no barrier needed: Y reads Sb/Mb/Ccb, this writes only S)
    float ec = __expf(sumA);
    #pragma unroll
    for (int tt=0;tt<4;tt++){
      int t = w*4 + tt, tm = (t>>3)*16, tn = (t&7)*16;
      v8f acc = {};
      for (int kb=0; kb<QC; kb+=32)
        acc = wmma_bf(frag_ld(&xqbT[(tm+l15)*SQ + kb + klo]),
                      frag_ld(&BdT [(tn+l15)*SQ + kb + klo]), acc);
      #pragma unroll
      for (int v=0;v<8;v++){
        int p = tm + v + ((lane>>4)<<3);
        int n = tn + l15;
        S[p*NST + n] = ec*S[p*NST + n] + acc[v];
      }
    }
  }
}

// --------------------------- y = rmsnorm(y_raw * silu(z)) * gw  -> bf16
__global__ __launch_bounds__(256)
void k_gate_norm(const float* __restrict__ y_raw, const float* __restrict__ proj,
                 const float* __restrict__ gw, bf16t* __restrict__ y_bf){
  int row = blockIdx.x;
  __shared__ float red[256];
  float loc[16]; float ss = 0.f;
  #pragma unroll
  for (int i=0;i<16;i++){
    int c = threadIdx.x + i*256;
    float z = proj[(size_t)row*NPROJ + c];
    float g = y_raw[(size_t)row*DIN_ + c] * siluf(z);
    loc[i] = g; ss += g*g;
  }
  red[threadIdx.x] = ss; __syncthreads();
  for (int s=128;s>0;s>>=1){ if (threadIdx.x < s) red[threadIdx.x] += red[threadIdx.x+s]; __syncthreads(); }
  float rstd = rsqrtf(red[0]/(float)DIN_ + EPSV);
  #pragma unroll
  for (int i=0;i<16;i++){
    int c = threadIdx.x + i*256;
    y_bf[(size_t)row*DIN_ + c] = (bf16t)(loc[i]*rstd*gw[c]);
  }
}

constexpr size_t alignup(size_t x){ return (x + 255) & ~(size_t)255; }

} // namespace

extern "C" void kernel_launch(void* const* d_in, const int* in_sizes, int n_in,
                              void* d_out, int out_size, void* d_ws, size_t ws_size,
                              hipStream_t stream){
  (void)in_sizes; (void)n_in; (void)out_size; (void)ws_size;
  const float* hs    = (const float*)d_in[0];
  const float* res   = (const float*)d_in[1];
  const float* nw    = (const float*)d_in[2];
  const float* w_in  = (const float*)d_in[3];
  const float* cw    = (const float*)d_in[4];
  const float* cb    = (const float*)d_in[5];
  const float* alog  = (const float*)d_in[6];
  const float* dtb   = (const float*)d_in[7];
  const float* dsk   = (const float*)d_in[8];
  const float* gw    = (const float*)d_in[9];
  const float* w_out = (const float*)d_in[10];

  char* ws = (char*)d_ws;
  constexpr size_t OFF_WIN  = 0;
  constexpr size_t OFF_WOUT = alignup(OFF_WIN  + (size_t)DDIM*NPROJ*sizeof(bf16t));
  constexpr size_t OFF_HBF  = alignup(OFF_WOUT + (size_t)DIN_*DDIM*sizeof(bf16t));
  constexpr size_t OFF_PROJ = alignup(OFF_HBF  + (size_t)ROWS*DDIM*sizeof(bf16t));
  constexpr size_t OFF_XBC  = alignup(OFF_PROJ + (size_t)ROWS*NPROJ*sizeof(float));
  constexpr size_t OFF_DTS  = alignup(OFF_XBC  + (size_t)ROWS*CONVD*sizeof(float));
  constexpr size_t OFF_YRAW = alignup(OFF_DTS  + (size_t)ROWS*HH*sizeof(float));
  constexpr size_t OFF_YBF  = alignup(OFF_YRAW + (size_t)ROWS*DIN_*sizeof(float));

  bf16t* w_in_bf  = (bf16t*)(ws + OFF_WIN);
  bf16t* w_out_bf = (bf16t*)(ws + OFF_WOUT);
  bf16t* h_bf     = (bf16t*)(ws + OFF_HBF);
  float* proj     = (float*)(ws + OFF_PROJ);
  float* xBCs     = (float*)(ws + OFF_XBC);
  float* dts      = (float*)(ws + OFF_DTS);
  float* y_raw    = (float*)(ws + OFF_YRAW);
  bf16t* y_bf     = (bf16t*)(ws + OFF_YBF);

  float* out       = (float*)d_out;
  float* resid_out = out + (size_t)ROWS*DDIM;

  k_f2bf<<<4096, 256, 0, stream>>>(w_in,  w_in_bf,  DDIM*NPROJ);
  k_f2bf<<<4096, 256, 0, stream>>>(w_out, w_out_bf, DIN_*DDIM);

  k_fuse_norm<<<ROWS, 256, 0, stream>>>(hs, res, nw, resid_out, h_bf);

  k_gemm_bf16<<<dim3((NPROJ + 127)/128, ROWS/128), 256, 0, stream>>>(
      h_bf, w_in_bf, proj, ROWS, NPROJ, DDIM);

  k_conv_silu<<<ROWS, 256, 0, stream>>>(proj, cw, cb, dtb, xBCs, dts);

  k_ssd<<<BDIM*HH, 256, 0, stream>>>(xBCs, dts, alog, dsk, y_raw);

  k_gate_norm<<<ROWS, 256, 0, stream>>>(y_raw, proj, gw, y_bf);

  k_gemm_bf16<<<dim3((DDIM + 127)/128, ROWS/128), 256, 0, stream>>>(
      y_bf, w_out_bf, out, ROWS, DDIM, DIN_);
}
